// GIN_28080496181806
// MI455X (gfx1250) — matrix operand order repaired and checked by reading, not compiled, not run on hardware
//
#include <hip/hip_runtime.h>

// GINE GNN forward for MI455X (gfx1250, wave32).
// GEMMs: v_wmma_f32_16x16x32_bf16 with weight B-fragments register-cached per
// wave and grid-stride tile loops. Message passing recomputes the edge
// projection (102 MB/layer HBM vs 410 MB/layer for materialized e) and
// scatter-adds into an L2-resident aggregation buffer with f32 atomics.

typedef __attribute__((ext_vector_type(16))) __bf16 v16bf;
typedef __attribute__((ext_vector_type(8)))  float  v8f;

#define N_NODES 50000
#define N_EDGES 1600000
#define DDIM    64
#define EINDIM  16
#define NBATCH  256
#define GDEPTH  3
#define LDSLD   68   // 16x64 f32 tile row stride (4*68B rows stay 16B aligned)
#define WPB     4    // waves per block in WMMA kernels
#define WMMA_BLOCKS 128

// ---- WMMA fragment helpers (layouts per CDNA5 ISA 7.12.2) ------------------

// A (16x32 bf16) from an LDS f32 tile with row stride LDSLD.
// lanes 0-15: row M=lane, K = {kb+0..7, kb+16..23}
// lanes 16-31: row M=lane-16, K = {kb+8..15, kb+24..31}
__device__ inline v16bf load_A_frag_lds(const float* A, int kbase, int lane) {
    const int m  = lane & 15;
    const int kh = (lane >> 4) & 1;
    const float* row = A + m * LDSLD + kbase + kh * 8;
    v16bf a;
#pragma unroll
    for (int j = 0; j < 8; ++j) {
        a[j]     = (__bf16)row[j];
        a[j + 8] = (__bf16)row[16 + j];
    }
    return a;
}

// B (32x16 bf16) from a row-major [64][64] f32 weight matrix.
// lanes 0-15: col N=nbase+lane, K = kbase+{0..7,16..23} ; lanes 16-31: K = kbase+{8..15,24..31}
__device__ inline v16bf load_B_frag(const float* __restrict__ W, int kbase, int nbase, int lane) {
    const int n  = nbase + (lane & 15);
    const int k0 = kbase + ((lane >> 4) & 1) * 8;
    v16bf b;
#pragma unroll
    for (int j = 0; j < 16; ++j) b[j] = (__bf16)W[(k0 + (j & 7) + (j >> 3) * 16) * DDIM + n];
    return b;
}

__device__ inline v8f wmma_bf16(v16bf a, v16bf b, v8f c) {
    return __builtin_amdgcn_wmma_f32_16x16x32_bf16(false, a, false, b, (short)0, c,
                                                   false, false);
}

// ---- Kernel 1: h = relu(x @ Wn + bn) ---------------------------------------

__global__ __launch_bounds__(WPB * 32)
void node_proj_kernel(const float* __restrict__ x, const float* __restrict__ Wn,
                      const float* __restrict__ bn, float* __restrict__ h, int nrows) {
    __shared__ float smem[WPB][16 * LDSLD];
    const int wave = threadIdx.x >> 5;
    const int lane = threadIdx.x & 31;
    const int ntiles = nrows / 16;
    const int nwaves = gridDim.x * WPB;
    float* A = smem[wave];
    const int ncol = lane & 15;
    const int mb   = ((lane >> 4) & 1) * 8;

    // Register-cache all Wn B-fragments + biases once per wave.
    v16bf wB[2][4];
    float bias[4];
#pragma unroll
    for (int nt = 0; nt < 4; ++nt) {
        wB[0][nt] = load_B_frag(Wn, 0,  nt * 16, lane);
        wB[1][nt] = load_B_frag(Wn, 32, nt * 16, lane);
        bias[nt]  = bn[nt * 16 + ncol];
    }

    for (int tile = blockIdx.x * WPB + wave; tile < ntiles; tile += nwaves) {
        const int row0 = tile * 16;
        // Stage 16x64 f32 tile, coalesced. (wave-local LDS: DS ops are in-order)
#pragma unroll
        for (int i = lane; i < 16 * 16; i += 32) {
            const int r = i >> 4, c = (i & 15) << 2;
            *(float4*)(A + r * LDSLD + c) =
                *(const float4*)(x + (size_t)(row0 + r) * DDIM + c);
        }
        const v16bf a0 = load_A_frag_lds(A, 0, lane);
        const v16bf a1 = load_A_frag_lds(A, 32, lane);
#pragma unroll
        for (int nt = 0; nt < 4; ++nt) {
            v8f acc = {};
            acc = wmma_bf16(a0, wB[0][nt], acc);
            acc = wmma_bf16(a1, wB[1][nt], acc);
            float* dst = h + (size_t)row0 * DDIM + nt * 16 + ncol;
#pragma unroll
            for (int r = 0; r < 8; ++r)
                dst[(size_t)(mb + r) * DDIM] = fmaxf(acc[r] + bias[nt], 0.0f);
        }
    }
}

// ---- Kernel 2: edge messages: aggr[dst] += relu(h[src] + (attr@We + be)) ---

__global__ __launch_bounds__(256)
void msg_kernel(const float* __restrict__ h, const float* __restrict__ eattr,
                const int* __restrict__ eidx, const float* __restrict__ We,
                const float* __restrict__ be, float* __restrict__ aggr, int nedges) {
    __shared__ float sWe[EINDIM * DDIM];
    __shared__ float sbe[DDIM];
    for (int i = threadIdx.x; i < EINDIM * DDIM; i += 256) sWe[i] = We[i];
    if (threadIdx.x < DDIM) sbe[threadIdx.x] = be[threadIdx.x];
    __syncthreads();

    const int gid = blockIdx.x * 256 + threadIdx.x;   // E*8 threads, 8 cols each
    const int e = gid >> 3;
    if (e >= nedges) return;
    const int q   = (gid & 7) << 3;
    const int src = eidx[e];
    const int dst = eidx[nedges + e];

    const float4* ap = (const float4*)(eattr + (size_t)e * EINDIM);
    const float4 A0 = ap[0], A1 = ap[1], A2 = ap[2], A3 = ap[3];
    const float at[16] = {A0.x, A0.y, A0.z, A0.w, A1.x, A1.y, A1.z, A1.w,
                          A2.x, A2.y, A2.z, A2.w, A3.x, A3.y, A3.z, A3.w};
    const float4 h0 = *(const float4*)(h + (size_t)src * DDIM + q);
    const float4 h1 = *(const float4*)(h + (size_t)src * DDIM + q + 4);
    const float hvv[8] = {h0.x, h0.y, h0.z, h0.w, h1.x, h1.y, h1.z, h1.w};
    float* out = aggr + (size_t)dst * DDIM + q;
#pragma unroll
    for (int c = 0; c < 8; ++c) {
        float acc = sbe[q + c];
#pragma unroll
        for (int k = 0; k < 16; ++k) acc += at[k] * sWe[k * DDIM + q + c];
        atomicAdd(out + c, fmaxf(hvv[c] + acc, 0.0f));
    }
}

// ---- Kernel 3: h' = [relu] ( relu((h+aggr)@W1 + b1) @ W2 + b2 ) ------------

__global__ __launch_bounds__(WPB * 32)
void mlp_kernel(const float* __restrict__ h, const float* __restrict__ aggr,
                const float* __restrict__ W1, const float* __restrict__ b1,
                const float* __restrict__ W2, const float* __restrict__ b2,
                float* __restrict__ hout, int nrows, int relu_out) {
    __shared__ float smemA[WPB][16 * LDSLD];
    __shared__ float smemT[WPB][16 * LDSLD];
    const int wave = threadIdx.x >> 5;
    const int lane = threadIdx.x & 31;
    const int ntiles = nrows / 16;
    const int nwaves = gridDim.x * WPB;
    float* A = smemA[wave];
    float* T = smemT[wave];
    const int ncol = lane & 15;
    const int mb   = ((lane >> 4) & 1) * 8;

    // Register-cache all W1/W2 B-fragments + biases once per wave.
    v16bf w1B[2][4], w2B[2][4];
    float bias1[4], bias2[4];
#pragma unroll
    for (int nt = 0; nt < 4; ++nt) {
        w1B[0][nt] = load_B_frag(W1, 0,  nt * 16, lane);
        w1B[1][nt] = load_B_frag(W1, 32, nt * 16, lane);
        w2B[0][nt] = load_B_frag(W2, 0,  nt * 16, lane);
        w2B[1][nt] = load_B_frag(W2, 32, nt * 16, lane);
        bias1[nt]  = b1[nt * 16 + ncol];
        bias2[nt]  = b2[nt * 16 + ncol];
    }

    for (int tile = blockIdx.x * WPB + wave; tile < ntiles; tile += nwaves) {
        const int row0 = tile * 16;
        // Stage u = h + aggr (16x64 f32), coalesced.
#pragma unroll
        for (int i = lane; i < 16 * 16; i += 32) {
            const int r = i >> 4, c = (i & 15) << 2;
            const size_t off = (size_t)(row0 + r) * DDIM + c;
            const float4 hv = *(const float4*)(h + off);
            const float4 av = *(const float4*)(aggr + off);
            float4 u;
            u.x = hv.x + av.x; u.y = hv.y + av.y;
            u.z = hv.z + av.z; u.w = hv.w + av.w;
            *(float4*)(A + r * LDSLD + c) = u;
        }

        // GEMM 1: T = relu(u @ W1 + b1)
        {
            const v16bf a0 = load_A_frag_lds(A, 0, lane);
            const v16bf a1 = load_A_frag_lds(A, 32, lane);
#pragma unroll
            for (int nt = 0; nt < 4; ++nt) {
                v8f acc = {};
                acc = wmma_bf16(a0, w1B[0][nt], acc);
                acc = wmma_bf16(a1, w1B[1][nt], acc);
#pragma unroll
                for (int r = 0; r < 8; ++r)
                    T[(mb + r) * LDSLD + nt * 16 + ncol] =
                        fmaxf(acc[r] + bias1[nt], 0.0f);
            }
        }

        // GEMM 2: hout = (T @ W2 + b2), optional relu. Same-wave LDS: in-order.
        {
            const v16bf t0 = load_A_frag_lds(T, 0, lane);
            const v16bf t1 = load_A_frag_lds(T, 32, lane);
#pragma unroll
            for (int nt = 0; nt < 4; ++nt) {
                v8f acc = {};
                acc = wmma_bf16(t0, w2B[0][nt], acc);
                acc = wmma_bf16(t1, w2B[1][nt], acc);
                float* dst = hout + (size_t)row0 * DDIM + nt * 16 + ncol;
#pragma unroll
                for (int r = 0; r < 8; ++r) {
                    float v = acc[r] + bias2[nt];
                    if (relu_out) v = fmaxf(v, 0.0f);
                    dst[(size_t)(mb + r) * DDIM] = v;
                }
            }
        }
    }
}

// ---- Kernel 4: readout[b] += h[n] for batch[n]==b --------------------------

__global__ __launch_bounds__(256)
void readout_kernel(const float* __restrict__ h, const int* __restrict__ batch,
                    float* __restrict__ out, int nrows) {
    const int gid = blockIdx.x * 256 + threadIdx.x;
    const int n = gid >> 4;
    if (n >= nrows) return;
    const int q = (gid & 15) << 2;
    const int b = batch[n];
    const float4 v = *(const float4*)(h + (size_t)n * DDIM + q);
    float* o = out + (size_t)b * DDIM + q;
    atomicAdd(o + 0, v.x);
    atomicAdd(o + 1, v.y);
    atomicAdd(o + 2, v.z);
    atomicAdd(o + 3, v.w);
}

__global__ void zero_kernel(float* __restrict__ p, int n) {
    const int i4 = (blockIdx.x * blockDim.x + threadIdx.x) << 2;
    if (i4 < n) *(float4*)(p + i4) = make_float4(0.f, 0.f, 0.f, 0.f);
}

// ---- Launch ----------------------------------------------------------------

extern "C" void kernel_launch(void* const* d_in, const int* in_sizes, int n_in,
                              void* d_out, int out_size, void* d_ws, size_t ws_size,
                              hipStream_t stream) {
    const float* x     = (const float*)d_in[0];
    const float* eattr = (const float*)d_in[1];
    const int*   eidx  = (const int*)d_in[2];
    const int*   batch = (const int*)d_in[3];
    const float* Wn = (const float*)d_in[4];
    const float* bn = (const float*)d_in[5];
    const float* We = (const float*)d_in[6];
    const float* be = (const float*)d_in[7];
    const float* W1 = (const float*)d_in[8];
    const float* b1 = (const float*)d_in[9];
    const float* W2 = (const float*)d_in[10];
    const float* b2 = (const float*)d_in[11];
    float* out = (float*)d_out;

    float* hA   = (float*)d_ws;
    float* hB   = hA + (size_t)N_NODES * DDIM;
    float* aggr = hB + (size_t)N_NODES * DDIM;

    const dim3 wblk(WPB * 32);

    node_proj_kernel<<<WMMA_BLOCKS, wblk, 0, stream>>>(x, Wn, bn, hA, N_NODES);

    float* hcur = hA;
    float* hnext = hB;
    for (int l = 0; l < GDEPTH; ++l) {
        zero_kernel<<<(N_NODES * DDIM / 4 + 255) / 256, 256, 0, stream>>>(aggr, N_NODES * DDIM);
        const int mgrid = (N_EDGES * 8 + 255) / 256;
        msg_kernel<<<mgrid, 256, 0, stream>>>(hcur, eattr, eidx, We, be, aggr, N_EDGES);
        mlp_kernel<<<WMMA_BLOCKS, wblk, 0, stream>>>(hcur, aggr,
                                                     W1 + (size_t)l * DDIM * DDIM, b1 + (size_t)l * DDIM,
                                                     W2 + (size_t)l * DDIM * DDIM, b2 + (size_t)l * DDIM,
                                                     hnext, N_NODES, (l < GDEPTH - 1) ? 1 : 0);
        float* t = hcur; hcur = hnext; hnext = t;
    }

    zero_kernel<<<(NBATCH * DDIM / 4 + 255) / 256, 256, 0, stream>>>(out, NBATCH * DDIM);
    readout_kernel<<<(N_NODES * 16 + 255) / 256, 256, 0, stream>>>(hcur, batch, out, N_NODES);
}
